// MoIETransformerBlock_4956392260099
// MI455X (gfx1250) — matrix operand order repaired and checked by reading, not compiled, not run on hardware
//
#include <hip/hip_runtime.h>
#include <math.h>

// ---------------------------------------------------------------------------
// Types for CDNA5 WMMA (gfx1250, wave32)
// ---------------------------------------------------------------------------
typedef __bf16 bf16;
typedef __attribute__((ext_vector_type(16))) __bf16 v16bf;
typedef __attribute__((ext_vector_type(8)))  __bf16 v8bf;
typedef __attribute__((ext_vector_type(8)))  float  v8f;

union V16 { v16bf v; v8bf h[2]; };

// ---------------------------------------------------------------------------
// Problem constants (from reference: B=4, S=2048, D=2048)
// ---------------------------------------------------------------------------
constexpr int    BATCH = 4;
constexpr int    SEQ   = 2048;
constexpr int    DIM   = 2048;
constexpr size_t DD    = (size_t)DIM * DIM;     // per-module weight matrix
constexpr size_t SD    = (size_t)SEQ * DIM;     // per-batch activation slab
constexpr size_t SS    = (size_t)SEQ * SEQ;     // per-batch score matrix
constexpr size_t BSD   = (size_t)BATCH * SD;

// Workspace layout (bytes). scores aliases prc (disjoint lifetimes).
constexpr size_t OFF_MUW16 = 0;                          // bf16 [4,D,D]
constexpr size_t OFF_PTW16 = OFF_MUW16 + 4 * DD * 2;     // bf16 [4,D,D]
constexpr size_t OFF_INC16 = OFF_PTW16 + 4 * DD * 2;     // bf16 [4,D,D]
constexpr size_t OFF_PROTO = OFF_INC16 + 4 * DD * 2;     // f32  [4,D,D] proto_state
constexpr size_t OFF_LN1   = OFF_PROTO + 4 * DD * 4;     // bf16 [B,S,D]
constexpr size_t OFF_COMP  = OFF_LN1   + BSD * 2;        // bf16 [3,B,S,D] q,k,v (masked)
constexpr size_t OFF_VT    = OFF_COMP  + 3 * BSD * 2;    // bf16 [B,D,S]
constexpr size_t OFF_PROBS = OFF_VT    + BSD * 2;        // bf16 [B,S,S]
constexpr size_t OFF_ATTN  = OFF_PROBS + (size_t)BATCH * SS * 2; // bf16 [B,S,D]
constexpr size_t OFF_BIG   = OFF_ATTN  + BSD * 2;        // f32  max(4*D*D, B*S*S) (prc, then scores)
constexpr size_t OFF_CTX   = OFF_BIG   + 4 * DD * 4;     // f32  [B,D]
constexpr size_t OFF_CTXO  = OFF_CTX   + (size_t)BATCH * DIM * 4;
constexpr size_t OFF_ROUT  = OFF_CTXO  + (size_t)BATCH * DIM * 4; // f32 [4,B,D]
constexpr size_t OFF_GATE  = OFF_ROUT  + 4ull * BATCH * DIM * 4;  // f32 [4,B,D]

// ---------------------------------------------------------------------------
// Block reductions
// ---------------------------------------------------------------------------
__device__ __forceinline__ float blk_sum(float v, float* sh) {
  int t = threadIdx.x; sh[t] = v; __syncthreads();
#pragma unroll
  for (int s = 128; s > 0; s >>= 1) { if (t < s) sh[t] += sh[t + s]; __syncthreads(); }
  float r = sh[0]; __syncthreads(); return r;
}
__device__ __forceinline__ float blk_max(float v, float* sh) {
  int t = threadIdx.x; sh[t] = v; __syncthreads();
#pragma unroll
  for (int s = 128; s > 0; s >>= 1) { if (t < s) sh[t] = fmaxf(sh[t], sh[t + s]); __syncthreads(); }
  float r = sh[0]; __syncthreads(); return r;
}

// ---------------------------------------------------------------------------
// WMMA NT GEMM: C[M,N] = A[M,K] (row-major) * B[N,K]^T (row-major), bf16 in,
// f32 accumulate via v_wmma_f32_16x16x32_bf16.
// Block tile 128x128, 8 waves (4 along M x 2 along N), each wave 32x64
// (2x4 WMMA tiles -> 8 WMMAs per K-step fed by 12 ds_load_b128, with 4x
// B-fragment reuse of each A fragment). LDS staged with pad-40 stride to
// break bank conflicts. Epilogues (template):
//   0: store f32
//   1: bf16( silu(acc+bias[n]) * gate[n] )          (QKV + routing mask)
//   2: f32( n<=m ? acc*scale : -3e38 )              (causal scores)
//   3: bf16(acc)                                    (attn out)
//   4: f32( resid + gate[n]*silu(acc+bias[n]) )     (final output + residual)
// ---------------------------------------------------------------------------
constexpr int BM = 128, BN = 128, KT = 32, LDSTR = 40;

template <int EPI>
__global__ __launch_bounds__(256) void gemm_nt(
    const bf16* __restrict__ A, int lda,
    const bf16* __restrict__ Bp, int ldb,
    float* __restrict__ Cf, bf16* __restrict__ Cb, int ldc,
    int M, int N, int K,
    const float* __restrict__ bias,
    const float* __restrict__ gate_row,
    const float* __restrict__ resid,
    float scale)
{
  const int m0 = blockIdx.x * BM;
  const int n0 = blockIdx.y * BN;
  const int tid = threadIdx.x;

  if (EPI == 2 && n0 >= m0 + BM) {     // whole tile strictly above diagonal
    for (int idx = tid; idx < BM * BN; idx += 256) {
      int r = idx / BN, c = idx % BN;
      Cf[(size_t)(m0 + r) * ldc + (n0 + c)] = -3.0e38f;
    }
    return;
  }

  __shared__ alignas(16) bf16 As[BM * LDSTR];
  __shared__ alignas(16) bf16 Bs[BN * LDSTR];

  const int wave  = tid >> 5;
  const int lane  = tid & 31;
  const int wm    = (wave & 3) * 32;   // 4 waves along M, 32 rows each
  const int wn    = (wave >> 2) * 64;  // 2 waves along N, 64 cols each
  const int lrow  = lane & 15;
  const int lhalf = lane >> 4;

  v8f acc[2][4];
#pragma unroll
  for (int i = 0; i < 2; ++i)
#pragma unroll
    for (int j = 0; j < 4; ++j)
#pragma unroll
      for (int e = 0; e < 8; ++e) acc[i][j][e] = 0.0f;

  for (int k0 = 0; k0 < K; k0 += KT) {
    // Stage A tile 128x32 and B tile 128x32 (512 x v8bf each)
#pragma unroll
    for (int it = 0; it < 2; ++it) {
      int idx = tid + it * 256;
      int row = idx >> 2, col = (idx & 3) * 8;
      *(v8bf*)(&As[row * LDSTR + col]) =
          *(const v8bf*)(&A[(size_t)(m0 + row) * lda + k0 + col]);
      *(v8bf*)(&Bs[row * LDSTR + col]) =
          *(const v8bf*)(&Bp[(size_t)(n0 + row) * ldb + k0 + col]);
    }
    __syncthreads();

    // Fragments per ISA VGPR layouts (contiguous 16B loads, no transposes)
    V16 af[2], bfm[4];
#pragma unroll
    for (int i = 0; i < 2; ++i) {
      int am = (wm + i * 16 + lrow) * LDSTR;
      af[i].h[0] = *(const v8bf*)(&As[am + lhalf * 8]);
      af[i].h[1] = *(const v8bf*)(&As[am + 16 + lhalf * 8]);
    }
#pragma unroll
    for (int j = 0; j < 4; ++j) {
      int bn = (wn + j * 16 + lrow) * LDSTR;
      bfm[j].h[0] = *(const v8bf*)(&Bs[bn + lhalf * 8]);
      bfm[j].h[1] = *(const v8bf*)(&Bs[bn + 16 + lhalf * 8]);
    }
#pragma unroll
    for (int i = 0; i < 2; ++i)
#pragma unroll
      for (int j = 0; j < 4; ++j)
        acc[i][j] = __builtin_amdgcn_wmma_f32_16x16x32_bf16(
            false, af[i].v, false, bfm[j].v, (short)0, acc[i][j], false, false);
    __syncthreads();
  }

  // Epilogue. C/D layout: VGPR r -> row r + 8*lhalf; column = lrow.
#pragma unroll
  for (int i = 0; i < 2; ++i)
#pragma unroll
    for (int j = 0; j < 4; ++j)
#pragma unroll
      for (int r = 0; r < 8; ++r) {
        int gm = m0 + wm + i * 16 + r + 8 * lhalf;
        int gn = n0 + wn + j * 16 + lrow;
        float v = acc[i][j][r];
        size_t o = (size_t)gm * ldc + gn;
        if (EPI == 0) {
          Cf[o] = v;
        } else if (EPI == 1) {
          float z = v + bias[gn];
          float sl = z / (1.0f + __expf(-z));
          Cb[o] = (bf16)(sl * gate_row[gn]);
        } else if (EPI == 2) {
          Cf[o] = (gn <= gm) ? v * scale : -3.0e38f;
        } else if (EPI == 3) {
          Cb[o] = (bf16)v;
        } else {
          float z = v + bias[gn];
          float sl = z / (1.0f + __expf(-z));
          Cf[o] = resid[o] + gate_row[gn] * sl;
        }
      }
}

// ---------------------------------------------------------------------------
// Elementwise / reduction helper kernels
// ---------------------------------------------------------------------------
__global__ void cvt_bf16_kernel(const float* __restrict__ src,
                                bf16* __restrict__ dst, size_t n) {
  size_t i = (size_t)blockIdx.x * blockDim.x + threadIdx.x;
  size_t stride = (size_t)gridDim.x * blockDim.x;
  for (; i < n; i += stride) dst[i] = (bf16)src[i];
}

// LN over rows of x, write bf16
__global__ __launch_bounds__(256) void ln_rows_kernel(
    const float* __restrict__ xin, const float* __restrict__ g,
    const float* __restrict__ bb, bf16* __restrict__ out, int D) {
  __shared__ float sh[256];
  size_t row = blockIdx.x;
  const float* xr = xin + row * D;
  float s1 = 0.f, s2 = 0.f;
  for (int d = threadIdx.x; d < D; d += 256) { float v = xr[d]; s1 += v; s2 += v * v; }
  s1 = blk_sum(s1, sh); s2 = blk_sum(s2, sh);
  float mean = s1 / D, var = s2 / D - mean * mean;
  float rstd = rsqrtf(var + 1e-5f);
  for (int d = threadIdx.x; d < D; d += 256)
    out[row * D + d] = (bf16)((xr[d] - mean) * rstd * g[d] + bb[d]);
}

// proto_state[n,i,:] = proto_w[n,i,:] + LN(prc[n,i,:]) * pln_g[n,:] + pln_b[n,:]
__global__ __launch_bounds__(256) void proto_ln_kernel(
    const float* __restrict__ prc, const float* __restrict__ proto_w,
    const float* __restrict__ pg, const float* __restrict__ pb,
    float* __restrict__ out, int D) {
  __shared__ float sh[256];
  size_t row = blockIdx.x;           // n*D + i
  int n = (int)(row / D);
  const float* pr = prc + row * D;
  float s1 = 0.f, s2 = 0.f;
  for (int d = threadIdx.x; d < D; d += 256) { float v = pr[d]; s1 += v; s2 += v * v; }
  s1 = blk_sum(s1, sh); s2 = blk_sum(s2, sh);
  float mean = s1 / D, var = s2 / D - mean * mean;
  float rstd = rsqrtf(var + 1e-5f);
  for (int d = threadIdx.x; d < D; d += 256)
    out[row * D + d] = proto_w[row * D + d] +
                       ((pr[d] - mean) * rstd) * pg[(size_t)n * D + d] + pb[(size_t)n * D + d];
}

// Column mean over S of a [B,S,D] bf16 tensor -> f32 [B,D]
__global__ void colmean_kernel(const bf16* __restrict__ a,
                               float* __restrict__ out, int S, int D) {
  int b = blockIdx.y;
  int d = blockIdx.x * blockDim.x + threadIdx.x;
  const bf16* p = a + (size_t)b * S * D + d;
  float s = 0.f;
  for (int i = 0; i < S; ++i) s += (float)p[(size_t)i * D];
  out[(size_t)b * D + d] = s / (float)S;
}

// routing[n,b,o] = (ctx[b,:] . proto_state[n,o,:]) * scale + gate[n,o]
// one wave per output element; coalesced lane-strided dot + shuffle reduce
__global__ __launch_bounds__(256) void routing_kernel(
    const float* __restrict__ ctxv, const float* __restrict__ proto_state,
    const float* __restrict__ gate, float* __restrict__ routing,
    int n_lo, int D, float scale) {
  int gw = blockIdx.x * 8 + (threadIdx.x >> 5);
  int lane = threadIdx.x & 31;
  int o = gw % D;
  int b = (gw / D) % BATCH;
  int n = n_lo + gw / (D * BATCH);
  const float* pr = proto_state + ((size_t)n * D + o) * D;
  const float* cv = ctxv + (size_t)b * D;
  float s = 0.f;
  for (int j = lane; j < D; j += 32) s += pr[j] * cv[j];
#pragma unroll
  for (int off = 16; off > 0; off >>= 1) s += __shfl_down(s, off, 32);
  if (lane == 0)
    routing[((size_t)n * BATCH + b) * D + o] = s * scale + gate[(size_t)n * D + o];
}

// gating = routing / (max|routing| + 1e-9) over last axis
__global__ __launch_bounds__(256) void gating_kernel(
    const float* __restrict__ routing, float* __restrict__ gating,
    int n_lo, int D) {
  __shared__ float sh[256];
  int nb = blockIdx.x;
  int n = n_lo + nb / BATCH, b = nb % BATCH;
  const float* r = routing + ((size_t)n * BATCH + b) * D;
  float mx = 0.f;
  for (int d = threadIdx.x; d < D; d += 256) mx = fmaxf(mx, fabsf(r[d]));
  mx = blk_max(mx, sh);
  float inv = 1.0f / (mx + 1e-9f);
  float* g = gating + ((size_t)n * BATCH + b) * D;
  for (int d = threadIdx.x; d < D; d += 256) g[d] = r[d] * inv;
}

// In-place RoPE over full hidden dim: pairs (d, d+D/2), one thread per pair
__global__ void rope_kernel(bf16* __restrict__ q, const int* __restrict__ pos,
                            int total_half, int S, int D) {
  int idx = blockIdx.x * blockDim.x + threadIdx.x;
  if (idx >= total_half) return;
  int half = D >> 1;
  int d = idx % half;
  int bs = idx / half;
  int s = bs % S;
  float t = (float)pos[s];
  float fr = powf(10000.0f, -((float)(2 * d) / (float)D));
  float ang = t * fr;
  float sn = sinf(ang), cs = cosf(ang);
  size_t base = (size_t)bs * D;
  float x0 = (float)q[base + d], x1 = (float)q[base + d + half];
  q[base + d]        = (bf16)(x0 * cs - x1 * sn);
  q[base + d + half] = (bf16)(x1 * cs + x0 * sn);
}

// bf16 transpose [B,S,D] -> [B,D,S] via padded LDS tile
__global__ __launch_bounds__(256) void transpose_kernel(
    const bf16* __restrict__ v, bf16* __restrict__ vt, int S, int D) {
  __shared__ bf16 tile[32][33];
  int b = blockIdx.z;
  int s0 = blockIdx.x * 32, d0 = blockIdx.y * 32;
  int tx = threadIdx.x & 31, ty = threadIdx.x >> 5;
  for (int i = ty; i < 32; i += 8)
    tile[i][tx] = v[((size_t)b * S + s0 + i) * D + d0 + tx];
  __syncthreads();
  for (int i = ty; i < 32; i += 8)
    vt[((size_t)b * D + d0 + i) * S + s0 + tx] = tile[tx][i];
}

// Row softmax over f32 scores (masked entries are -3e38 -> exp underflows to 0)
__global__ __launch_bounds__(256) void softmax_kernel(
    const float* __restrict__ sc, bf16* __restrict__ pr, int S) {
  __shared__ float sh[256];
  size_t row = blockIdx.x;
  const float* r = sc + row * S;
  float mx = -3.4e38f;
  for (int d = threadIdx.x; d < S; d += 256) mx = fmaxf(mx, r[d]);
  mx = blk_max(mx, sh);
  float sum = 0.f;
  for (int d = threadIdx.x; d < S; d += 256) sum += __expf(r[d] - mx);
  sum = blk_sum(sum, sh);
  float inv = 1.0f / sum;
  for (int d = threadIdx.x; d < S; d += 256)
    pr[row * S + d] = (bf16)(__expf(r[d] - mx) * inv);
}

// ---------------------------------------------------------------------------
// Orchestration
// ---------------------------------------------------------------------------
extern "C" void kernel_launch(void* const* d_in, const int* in_sizes, int n_in,
                              void* d_out, int out_size, void* d_ws, size_t ws_size,
                              hipStream_t stream) {
  (void)in_sizes; (void)n_in; (void)out_size; (void)ws_size;

  const float* x        = (const float*)d_in[0];
  const int*   pos      = (const int*)  d_in[1];
  const float* ln1_g    = (const float*)d_in[2];
  const float* ln1_b    = (const float*)d_in[3];
  const float* mu_w     = (const float*)d_in[4];
  const float* mu_b     = (const float*)d_in[5];
  const float* proto_w  = (const float*)d_in[6];
  const float* gatep    = (const float*)d_in[7];
  const float* pt_w     = (const float*)d_in[8];
  const float* pln_g    = (const float*)d_in[9];
  const float* pln_b    = (const float*)d_in[10];
  const float* incoming = (const float*)d_in[11];
  float* outp = (float*)d_out;

  char* ws = (char*)d_ws;
  bf16*  muw16  = (bf16*)(ws + OFF_MUW16);
  bf16*  ptw16  = (bf16*)(ws + OFF_PTW16);
  bf16*  inc16  = (bf16*)(ws + OFF_INC16);
  float* protoS = (float*)(ws + OFF_PROTO);
  bf16*  ln1    = (bf16*)(ws + OFF_LN1);
  bf16*  comp   = (bf16*)(ws + OFF_COMP);   // [3,B,S,D]: q,k,v
  bf16*  vT     = (bf16*)(ws + OFF_VT);
  bf16*  probs  = (bf16*)(ws + OFF_PROBS);
  bf16*  attn   = (bf16*)(ws + OFF_ATTN);
  float* big    = (float*)(ws + OFF_BIG);   // prc, then scores
  float* ctx    = (float*)(ws + OFF_CTX);
  float* ctx_o  = (float*)(ws + OFF_CTXO);
  float* routing= (float*)(ws + OFF_ROUT);
  float* gating = (float*)(ws + OFF_GATE);

  const float scale = 1.0f / sqrtf((float)DIM);

  // 1) fp32 -> bf16 weight copies
  cvt_bf16_kernel<<<4096, 256, 0, stream>>>(mu_w,     muw16, 4 * DD);
  cvt_bf16_kernel<<<4096, 256, 0, stream>>>(pt_w,     ptw16, 4 * DD);
  cvt_bf16_kernel<<<4096, 256, 0, stream>>>(incoming, inc16, 4 * DD);

  // 2) prc[n] = incoming[n] @ pt_w[n]^T  (NT, f32 out into `big`)
  for (int n = 0; n < 4; ++n)
    gemm_nt<0><<<dim3(DIM / BM, DIM / BN), 256, 0, stream>>>(
        inc16 + (size_t)n * DD, DIM, ptw16 + (size_t)n * DD, DIM,
        big + (size_t)n * DD, nullptr, DIM, DIM, DIM, DIM,
        nullptr, nullptr, nullptr, 0.f);

  // 3) proto_state = proto_w + LN(prc)
  proto_ln_kernel<<<4 * DIM, 256, 0, stream>>>(big, proto_w, pln_g, pln_b, protoS, DIM);

  // 4) ln1_out = LN(x); ctx = mean_s(ln1_out)
  ln_rows_kernel<<<BATCH * SEQ, 256, 0, stream>>>(x, ln1_g, ln1_b, ln1, DIM);
  colmean_kernel<<<dim3(DIM / 256, BATCH), 256, 0, stream>>>(ln1, ctx, SEQ, DIM);

  // 5) routing + MAS gating for q,k,v modules
  routing_kernel<<<(3 * BATCH * DIM) / 8, 256, 0, stream>>>(ctx, protoS, gatep, routing, 0, DIM, scale);
  gating_kernel<<<3 * BATCH, 256, 0, stream>>>(routing, gating, 0, DIM);

  // 6) q,k,v = gating * silu(ln1_out @ mu_w[n]^T + mu_b[n])
  for (int n = 0; n < 3; ++n)
    for (int b = 0; b < BATCH; ++b)
      gemm_nt<1><<<dim3(SEQ / BM, DIM / BN), 256, 0, stream>>>(
          ln1 + (size_t)b * SD, DIM, muw16 + (size_t)n * DD, DIM,
          nullptr, comp + ((size_t)n * BATCH + b) * SD, DIM, SEQ, DIM, DIM,
          mu_b + (size_t)n * DIM, gating + ((size_t)n * BATCH + b) * DIM, nullptr, 0.f);

  // 7) RoPE on q and k (in place)
  {
    int half_total = BATCH * SEQ * (DIM / 2);
    int blks = (half_total + 255) / 256;
    rope_kernel<<<blks, 256, 0, stream>>>(comp,        pos, half_total, SEQ, DIM); // q
    rope_kernel<<<blks, 256, 0, stream>>>(comp + BSD,  pos, half_total, SEQ, DIM); // k
  }

  // 8) vT = transpose(v) so probs@v is NT too
  transpose_kernel<<<dim3(SEQ / 32, DIM / 32, BATCH), 256, 0, stream>>>(
      comp + 2 * BSD, vT, SEQ, DIM);

  // 9) scores = causal_mask(q @ k^T * scale)   (reuses `big`)
  for (int b = 0; b < BATCH; ++b)
    gemm_nt<2><<<dim3(SEQ / BM, SEQ / BN), 256, 0, stream>>>(
        comp + (size_t)b * SD, DIM, comp + BSD + (size_t)b * SD, DIM,
        big + (size_t)b * SS, nullptr, SEQ, SEQ, SEQ, DIM,
        nullptr, nullptr, nullptr, scale);

  // 10) probs = softmax(scores)
  softmax_kernel<<<BATCH * SEQ, 256, 0, stream>>>(big, probs, SEQ);

  // 11) attn_out = probs @ v  (as probs @ vT^T, NT)
  for (int b = 0; b < BATCH; ++b)
    gemm_nt<3><<<dim3(SEQ / BM, DIM / BN), 256, 0, stream>>>(
        probs + (size_t)b * SS, SEQ, vT + (size_t)b * SD, SEQ,
        nullptr, attn + (size_t)b * SD, DIM, SEQ, DIM, SEQ,
        nullptr, nullptr, nullptr, 0.f);

  // 12) output routing from ctx_o
  colmean_kernel<<<dim3(DIM / 256, BATCH), 256, 0, stream>>>(attn, ctx_o, SEQ, DIM);
  routing_kernel<<<(BATCH * DIM) / 8, 256, 0, stream>>>(ctx_o, protoS, gatep, routing, 3, DIM, scale);
  gating_kernel<<<BATCH, 256, 0, stream>>>(routing, gating, 3, DIM);

  // 13) out = x + gating_o * silu(attn_out @ mu_w[3]^T + mu_b[3])
  for (int b = 0; b < BATCH; ++b)
    gemm_nt<4><<<dim3(SEQ / BM, DIM / BN), 256, 0, stream>>>(
        attn + (size_t)b * SD, DIM, muw16 + 3 * DD, DIM,
        outp + (size_t)b * SD, nullptr, DIM, SEQ, DIM, DIM,
        mu_b + 3 * DIM, gating + ((size_t)3 * BATCH + b) * DIM, x + (size_t)b * SD, 0.f);
}
